// SelfAttention_65111704207780
// MI455X (gfx1250) — matrix operand order repaired and checked by reading, not compiled, not run on hardware
//
#include <hip/hip_runtime.h>
#include <hip/hip_bf16.h>

typedef __attribute__((ext_vector_type(16))) _Float16 v16h;
typedef __attribute__((ext_vector_type(8)))  _Float16 v8h;
typedef __attribute__((ext_vector_type(8)))  float    v8f;

#define SEQ   2048
#define NB    2
#define NH    16
#define DH    64
#define DMODEL 1024
#define MTOT  (NB*SEQ)

static __device__ __forceinline__ v16h cat8(v8h a, v8h b) {
  return __builtin_shufflevector(a, b, 0,1,2,3,4,5,6,7,8,9,10,11,12,13,14,15);
}
static __device__ __forceinline__ v8f wmma_f16(v16h a, v16h b, v8f c) {
  return __builtin_amdgcn_wmma_f32_16x16x32_f16(false, a, false, b, (short)0, c, false, false);
}

// Async copy 32 bytes (16 f16) from global to LDS, tracked by ASYNCcnt.
// INST_OFFSET is added to BOTH the LDS and the memory address (ISA 15.18.3),
// so one base pointer pair serves both 16-byte halves.
static __device__ __forceinline__ void async_cp32(const _Float16* g, unsigned ldsOff) {
  asm volatile("global_load_async_to_lds_b128 %0, %1, off\n\t"
               "global_load_async_to_lds_b128 %0, %1, off offset:16"
               :: "v"(ldsOff), "v"(g) : "memory");
}
static __device__ __forceinline__ void wait_async0() {
  asm volatile("s_wait_asynccnt 0" ::: "memory");
}

// ---------------- stage 0: casts ----------------
__global__ void cast_f32_f16(const float* __restrict__ src, _Float16* __restrict__ dst, int n) {
  int i = blockIdx.x * 256 + threadIdx.x;
  if (i < n) dst[i] = (_Float16)src[i];
}

// Wt[n][k] = (f16) W[k][n]   (W is [1024][1024] row-major, k=in, n=out)
__global__ void cast_transpose_w(const float* __restrict__ W, _Float16* __restrict__ Wt) {
  int i = blockIdx.x * 256 + threadIdx.x;   // 0 .. 1M-1
  int k = i >> 10, n = i & 1023;
  Wt[(size_t)n * DMODEL + k] = (_Float16)W[(size_t)k * DMODEL + n];
}

// ---------------- stage 1: GEMM  out = A[M,1024] x Bt[1024,1024]^T-layout + bias ----------------
// mode 0: f16 row-major, *scale      (Q with scale=0.125, K row-major)
// mode 1: f16 transposed per head -> Vt[b,h,d,s]
// mode 2: f32 row-major (final projection)
__global__ __launch_bounds__(256) void gemm_wmma(
    const _Float16* __restrict__ A,   // [M][1024]
    const _Float16* __restrict__ Bt,  // [1024(N)][1024(K)]
    const float*    __restrict__ bias,// [1024]
    void* __restrict__ outp,
    int mode, float scale)
{
  __shared__ __align__(16) _Float16 As[2][128][40];
  __shared__ __align__(16) _Float16 Bs[2][128][40];
  const int tid   = threadIdx.x;
  const int lane  = tid & 31, wave = tid >> 5;
  const int waveM = wave & 1, waveN = wave >> 1;     // 2 x 4 wave grid, each wave 64x32
  const int m0 = blockIdx.y * 128;
  const int n0 = blockIdx.x * 128;

  v8f acc[4][2];
  #pragma unroll
  for (int i = 0; i < 4; i++)
    #pragma unroll
    for (int j = 0; j < 2; j++) acc[i][j] = {};

  const int ldRow = tid >> 1;        // 0..127
  const int ldCol = (tid & 1) * 16;  // 0 or 16 halves
  const int aoff  = (lane >> 4) * 8;
  const int row   = lane & 15;
  const int boff  = (lane >> 4) * 8;

  const _Float16* gA = A  + (size_t)(m0 + ldRow) * DMODEL + ldCol;
  const _Float16* gB = Bt + (size_t)(n0 + ldRow) * DMODEL + ldCol;

  // Issue async DMA of one 128x32 K-slice of both tiles into buffer `buf`.
  auto issue_tiles = [&](int buf, int kk) {
    async_cp32(gA + kk, (unsigned)(uintptr_t)&As[buf][ldRow][ldCol]);
    async_cp32(gB + kk, (unsigned)(uintptr_t)&Bs[buf][ldRow][ldCol]);
  };

  issue_tiles(0, 0);            // prologue prefetch
  wait_async0();
  __syncthreads();

  int cur = 0;
  for (int kk = 0; kk < DMODEL; kk += 32) {
    if (kk + 32 < DMODEL) issue_tiles(cur ^ 1, kk + 32);   // prefetch next slice

    v16h af[4];
    #pragma unroll
    for (int fm = 0; fm < 4; fm++) {
      const _Float16* p = &As[cur][waveM * 64 + fm * 16 + row][0];
      af[fm] = cat8(*(const v8h*)(p + aoff), *(const v8h*)(p + 16 + aoff));
    }
    v16h bf[2];
    #pragma unroll
    for (int fn = 0; fn < 2; fn++) {
      const _Float16* p = &Bs[cur][waveN * 32 + fn * 16 + row][boff];
      bf[fn] = cat8(*(const v8h*)p, *(const v8h*)(p + 16));
    }
    #pragma unroll
    for (int fm = 0; fm < 4; fm++)
      #pragma unroll
      for (int fn = 0; fn < 2; fn++)
        acc[fm][fn] = wmma_f16(af[fm], bf[fn], acc[fm][fn]);

    wait_async0();              // next buffer landed
    __syncthreads();            // all waves done reading `cur`, DMA visible WGP-wide
    cur ^= 1;
  }

  // epilogue: C layout element (m = r + 8*(lane>>4), n = lane&15)
  const int nLane = lane & 15;
  const int mHalf = (lane >> 4) * 8;
  #pragma unroll
  for (int fm = 0; fm < 4; fm++) {
    #pragma unroll
    for (int fn = 0; fn < 2; fn++) {
      const int nG = n0 + waveN * 32 + fn * 16 + nLane;
      const float bv = bias[nG];
      #pragma unroll
      for (int r = 0; r < 8; r++) {
        const int mG = m0 + waveM * 64 + fm * 16 + mHalf + r;
        const float v = (acc[fm][fn][r] + bv) * scale;
        if (mode == 0) {
          ((_Float16*)outp)[(size_t)mG * DMODEL + nG] = (_Float16)v;
        } else if (mode == 1) {
          const int b = mG >> 11, s = mG & (SEQ - 1);
          const int h = nG >> 6,  d = nG & (DH - 1);
          ((_Float16*)outp)[(((size_t)(b * NH + h) * DH + d) << 11) + s] = (_Float16)v;
        } else {
          ((float*)outp)[(size_t)mG * DMODEL + nG] = v;
        }
      }
    }
  }
}

// ---------------- stage 2: causal flash attention ----------------
// grid = (S/64, B*H), block = 128 (4 waves x 16 q-rows)
__global__ __launch_bounds__(128) void flash_attn(
    const _Float16* __restrict__ Qh,  // [M][1024], pre-scaled by 1/8
    const _Float16* __restrict__ Kh,  // [M][1024]
    const _Float16* __restrict__ Vt,  // [B*H*64][2048]  (d-major, key-minor)
    _Float16* __restrict__ ctx)       // [M][1024]
{
  __shared__ __align__(16) _Float16 Pl[4][16][40];   // per-wave P relayout patch
  const int tid  = threadIdx.x;
  const int lane = tid & 31, wave = tid >> 5;
  const int bh = blockIdx.y;             // 0..31
  const int b  = bh >> 4, h = bh & 15;
  const int qw0 = blockIdx.x * 64 + wave * 16;   // wave's first q row in sequence
  const int hBase = h * DH;
  const size_t rowB = (size_t)b * SEQ;

  const int aoff  = (lane >> 4) * 8;
  const int arow  = lane & 15;
  const int nLane = lane & 15;
  const int boff  = (lane >> 4) * 8;
  const int mHalf = (lane >> 4) * 8;

  // Q fragments (A layout), d = 0..31 and 32..63
  v16h qa[2];
  {
    const _Float16* qp = Qh + (rowB + qw0 + arow) * DMODEL + hBase;
    #pragma unroll
    for (int f = 0; f < 2; f++)
      qa[f] = cat8(*(const v8h*)(qp + f * 32 + aoff),
                   *(const v8h*)(qp + f * 32 + 16 + aoff));
  }

  v8f o[4];
  #pragma unroll
  for (int f = 0; f < 4; f++) o[f] = {};
  float mrow[8], lrow[8];
  #pragma unroll
  for (int r = 0; r < 8; r++) { mrow[r] = -1e30f; lrow[r] = 0.0f; }

  const int nKK = (qw0 + 47) >> 5;   // key blocks of 32 needed for causal coverage

  for (int it = 0; it < nKK; ++it) {
    const int kk = it * 32;

    // S = Q K^T, two 16-key fragments
    v8f s[2];
    #pragma unroll
    for (int kh = 0; kh < 2; ++kh) {
      const _Float16* kp = Kh + (rowB + kk + kh * 16 + nLane) * DMODEL + hBase + boff;
      v16h kb0 = cat8(*(const v8h*)kp,        *(const v8h*)(kp + 16));
      v16h kb1 = cat8(*(const v8h*)(kp + 32), *(const v8h*)(kp + 48));
      v8f c = {};
      c = wmma_f16(qa[0], kb0, c);
      c = wmma_f16(qa[1], kb1, c);
      s[kh] = c;
    }

    // causal mask (reference masks scaled scores with -10000)
    #pragma unroll
    for (int kh = 0; kh < 2; ++kh) {
      const int key = kk + kh * 16 + nLane;
      #pragma unroll
      for (int r = 0; r < 8; ++r) {
        const int q = qw0 + mHalf + r;
        if (key > q) s[kh][r] = -10000.0f;
      }
    }

    // online softmax: row max / rescale
    float alpha[8];
    #pragma unroll
    for (int r = 0; r < 8; ++r) {
      float mx = fmaxf(s[0][r], s[1][r]);
      #pragma unroll
      for (int msk = 8; msk >= 1; msk >>= 1)
        mx = fmaxf(mx, __shfl_xor(mx, msk, 16));
      const float mnew = fmaxf(mrow[r], mx);
      alpha[r] = __expf(mrow[r] - mnew);
      mrow[r]  = mnew;
    }

    // P = exp(S - m) -> f16 into LDS (C layout write, A layout read)
    #pragma unroll
    for (int kh = 0; kh < 2; ++kh) {
      #pragma unroll
      for (int r = 0; r < 8; ++r) {
        const float p = __expf(s[kh][r] - mrow[r]);
        s[kh][r] = p;
        Pl[wave][mHalf + r][kh * 16 + nLane] = (_Float16)p;
      }
    }
    #pragma unroll
    for (int r = 0; r < 8; ++r) {
      float sum = s[0][r] + s[1][r];
      #pragma unroll
      for (int msk = 8; msk >= 1; msk >>= 1)
        sum += __shfl_xor(sum, msk, 16);
      lrow[r] = lrow[r] * alpha[r] + sum;
      #pragma unroll
      for (int f = 0; f < 4; f++) o[f][r] *= alpha[r];
    }

    asm volatile("s_wait_dscnt 0" ::: "memory");   // wave-local LDS RAW fence

    const _Float16* pp = &Pl[wave][arow][0];
    v16h pa = cat8(*(const v8h*)(pp + aoff), *(const v8h*)(pp + 16 + aoff));

    // O += P V, four 16-wide d fragments from transposed V
    #pragma unroll
    for (int f = 0; f < 4; ++f) {
      const _Float16* vp = Vt + ((size_t)bh * DH + f * 16 + nLane) * SEQ + kk + boff;
      v16h vb = cat8(*(const v8h*)vp, *(const v8h*)(vp + 16));
      o[f] = wmma_f16(pa, vb, o[f]);
    }
  }

  // normalize and store context (f16)
  #pragma unroll
  for (int r = 0; r < 8; ++r) {
    const float inv = 1.0f / lrow[r];
    #pragma unroll
    for (int f = 0; f < 4; f++) o[f][r] *= inv;
  }
  #pragma unroll
  for (int f = 0; f < 4; ++f) {
    #pragma unroll
    for (int r = 0; r < 8; ++r) {
      const int q = qw0 + mHalf + r;
      ctx[(rowB + q) * DMODEL + hBase + f * 16 + nLane] = (_Float16)o[f][r];
    }
  }
}

// ---------------- launcher ----------------
extern "C" void kernel_launch(void* const* d_in, const int* in_sizes, int n_in,
                              void* d_out, int out_size, void* d_ws, size_t ws_size,
                              hipStream_t stream) {
  const float* x  = (const float*)d_in[0];
  const float* Wq = (const float*)d_in[1];
  const float* bq = (const float*)d_in[2];
  const float* Wk = (const float*)d_in[3];
  const float* bk = (const float*)d_in[4];
  const float* Wv = (const float*)d_in[5];
  const float* bv = (const float*)d_in[6];
  const float* Wo = (const float*)d_in[7];
  const float* bo = (const float*)d_in[8];
  float* out = (float*)d_out;

  char* ws = (char*)d_ws;
  const size_t MB = 1024u * 1024u;
  _Float16* xh   = (_Float16*)(ws + 0);          // 8 MB  : x cast [4096][1024]
  _Float16* WqT  = (_Float16*)(ws + 8  * MB);    // 2 MB
  _Float16* WkT  = (_Float16*)(ws + 10 * MB);    // 2 MB
  _Float16* WvT  = (_Float16*)(ws + 12 * MB);    // 2 MB
  _Float16* WoT  = (_Float16*)(ws + 14 * MB);    // 2 MB
  _Float16* Qh   = (_Float16*)(ws + 16 * MB);    // 8 MB  (pre-scaled by 1/8)
  _Float16* Kh   = (_Float16*)(ws + 24 * MB);    // 8 MB
  _Float16* VtT  = (_Float16*)(ws + 32 * MB);    // 8 MB  [B,H,Dh,S]
  _Float16* ctxh = (_Float16*)(ws + 40 * MB);    // 8 MB

  const int nX = MTOT * DMODEL;                  // 4,194,304
  cast_f32_f16<<<(nX + 255) / 256, 256, 0, stream>>>(x, xh, nX);
  cast_transpose_w<<<4096, 256, 0, stream>>>(Wq, WqT);
  cast_transpose_w<<<4096, 256, 0, stream>>>(Wk, WkT);
  cast_transpose_w<<<4096, 256, 0, stream>>>(Wv, WvT);
  cast_transpose_w<<<4096, 256, 0, stream>>>(Wo, WoT);

  dim3 gGemm(DMODEL / 128, MTOT / 128);          // (8, 32)
  gemm_wmma<<<gGemm, 256, 0, stream>>>(xh, WqT, bq, Qh,  0, 0.125f); // Q, scale folded
  gemm_wmma<<<gGemm, 256, 0, stream>>>(xh, WkT, bk, Kh,  0, 1.0f);   // K
  gemm_wmma<<<gGemm, 256, 0, stream>>>(xh, WvT, bv, VtT, 1, 1.0f);   // V, transposed store

  flash_attn<<<dim3(SEQ / 64, NB * NH), 128, 0, stream>>>(Qh, Kh, VtT, ctxh);

  gemm_wmma<<<gGemm, 256, 0, stream>>>(ctxh, WoT, bo, out, 2, 1.0f); // final, fp32 out
}
